// LexMFVI_7730941132801
// MI455X (gfx1250) — matrix operand order repaired and checked by reading, not compiled, not run on hardware
//
#include <hip/hip_runtime.h>
#include <cstdint>

// LexMFVI fused MFVI (3 iters), B=8, L=160.
// Bandwidth-bound (~1.3 GB min traffic @ 23.3 TB/s). arc einsum uses
// v_wmma_f32_16x16x32_f16 (diag-of-product trick); f_sib tiles staged to LDS
// via global_load_async_to_lds_b128 (ASYNCcnt path).

#define LW   160
#define L2W  25600      // LW*LW
#define BW   8
#define FSP  164        // padded row stride (floats) for staged f_sib tile

typedef __attribute__((ext_vector_type(16))) _Float16 v16h;
typedef __attribute__((ext_vector_type(8)))  _Float16 v8h;
typedef __attribute__((ext_vector_type(8)))  float    v8f;

__device__ __forceinline__ float wred_max(float v) {
#pragma unroll
  for (int m = 16; m >= 1; m >>= 1) v = fmaxf(v, __shfl_xor(v, m, 32));
  return v;
}
__device__ __forceinline__ float wred_sum(float v) {
#pragma unroll
  for (int m = 16; m >= 1; m >>= 1) v += __shfl_xor(v, m, 32);
  return v;
}
__device__ __forceinline__ float sigmoidf(float x) { return 1.f / (1.f + __expf(-x)); }

// ---------------- init: q_arc = softmax(s_arc, -1), one wave per row ----------------
__global__ void row_softmax_k(const float* __restrict__ in, float* __restrict__ out) {
  int wid  = (blockIdx.x * blockDim.x + threadIdx.x) >> 5;
  int lane = threadIdx.x & 31;
  if (wid >= BW * LW) return;
  const float* row = in + (size_t)wid * LW;
  float x[5];
  float mx = -3.4e38f;
#pragma unroll
  for (int u = 0; u < 5; ++u) { x[u] = row[lane + 32 * u]; mx = fmaxf(mx, x[u]); }
  mx = wred_max(mx);
  float s = 0.f;
#pragma unroll
  for (int u = 0; u < 5; ++u) { x[u] = __expf(x[u] - mx); s += x[u]; }
  s = wred_sum(s);
  float inv = 1.f / s;
#pragma unroll
  for (int u = 0; u < 5; ++u) out[(size_t)wid * LW + lane + 32 * u] = x[u] * inv;
}

// ---------------- init: q_span^0 = sigmoid(roll(s_span, +1, axis=1)) ----------------
__global__ void init_qspan_k(const float* __restrict__ s_span, float* __restrict__ qs0) {
  int idx = blockIdx.x * blockDim.x + threadIdx.x;
  if (idx >= BW * L2W) return;
  int b = idx / L2W; int rem = idx - b * L2W; int l = rem / LW; int r = rem - l * LW;
  int lm1 = (l == 0) ? (LW - 1) : (l - 1);
  qs0[idx] = sigmoidf(s_span[((size_t)b * LW + lm1) * LW + r]);
}

// ---------------- span/head fused iteration: one wave per (b,l,r) row ----------------
// q_head^{t-1} = softmax_h(s_head + f_dsp*scale) recomputed in-register,
// span_logit = span_w + <f_dsp, q_head^{t-1}>, q_span^t = sigmoid(span_logit).
// mode 3 additionally writes out2 (rolled sigmoid) and out3 = s_head + f_dsp*q_span^2.
__global__ void span_iter_k(const float* __restrict__ s_span,
                            const float* __restrict__ s_head,
                            const float* __restrict__ f_dsp,
                            const float* __restrict__ scale_buf,   // q_span^{t-2} (null for mode 1)
                            const float* __restrict__ q2_buf,      // q_span^2 (mode 3 only)
                            float* __restrict__ qs_out,            // q_span^t (modes 1,2)
                            float* __restrict__ out2,
                            float* __restrict__ out3,
                            int mode) {
  int wid  = (blockIdx.x * blockDim.x + threadIdx.x) >> 5;
  int lane = threadIdx.x & 31;
  if (wid >= BW * L2W) return;
  const float* sh = s_head + (size_t)wid * LW;
  const float* fd = f_dsp + (size_t)wid * LW;
  float scale = (mode == 1) ? 0.f : scale_buf[wid];
  float sv[5], fv[5], xv[5];
  float mx = -3.4e38f;
#pragma unroll
  for (int u = 0; u < 5; ++u) {
    int h = lane + 32 * u;
    sv[u] = sh[h]; fv[u] = fd[h];
    xv[u] = sv[u] + fv[u] * scale;
    mx = fmaxf(mx, xv[u]);
  }
  mx = wred_max(mx);
  float se = 0.f, sd = 0.f;
#pragma unroll
  for (int u = 0; u < 5; ++u) { float e = __expf(xv[u] - mx); se += e; sd += fv[u] * e; }
  se = wred_sum(se);
  sd = wred_sum(sd);
  float msg = sd / se;

  int b = wid / L2W; int rem = wid - b * L2W; int l = rem / LW; int r = rem - l * LW;
  int lm1 = (l == 0) ? (LW - 1) : (l - 1);
  float logit = s_span[((size_t)b * LW + lm1) * LW + r] + msg;

  if (mode != 3) {
    if (lane == 0) qs_out[wid] = sigmoidf(logit);
  } else {
    if (lane == 0) out2[((size_t)b * LW + lm1) * LW + r] = sigmoidf(logit);
    float q2 = q2_buf[wid];
#pragma unroll
    for (int u = 0; u < 5; ++u)
      out3[(size_t)wid * LW + lane + 32 * u] = sv[u] + fv[u] * q2;
  }
}

// ---------------- arc iteration: q_out = softmax_j(s_arc + sum_k f_sib[i,j,k]*q_in[k,j]) --
// One block per (b,i), 160 threads (5 waves). q_in[b] is transposed to f16 in LDS (QT).
// Per 16-wide j-tile: f_sib rows async-staged to LDS, each wave runs one
// v_wmma_f32_16x16x32_f16 on its K-chunk of 32 and contributes the C-diagonal.
__global__ void __launch_bounds__(160) arc_iter_k(const float* __restrict__ s_arc,
                                                  const float* __restrict__ f_sib,
                                                  const float* __restrict__ q_in,
                                                  float* __restrict__ q_out) {
  __shared__ _Float16 QT[LW][LW];     // QT[j][k] = q_in[b][k][j] as f16 (51200 B)
  __shared__ float FS[16][FSP];       // staged f_sib[b,i,j0:j0+16,:]      (10496 B)
  __shared__ float tpart[5][LW];      // per-wave diagonal partials        ( 3200 B)
  __shared__ float red[10];

  int bi   = blockIdx.x;              // b*L + i
  int b    = bi / LW;
  int tid  = threadIdx.x;
  int wave = tid >> 5, lane = tid & 31;
  int m16  = lane & 15, hf = lane >> 4;

  // build transposed f16 copy of q_in[b] (coalesced global reads, LDS scatter)
  const float* qb = q_in + (size_t)b * L2W;
  for (int idx = tid; idx < L2W; idx += 160) {
    int k = idx / LW; int j = idx - k * LW;
    QT[j][k] = (_Float16)qb[idx];
  }

  const float* frow = f_sib + (size_t)bi * L2W;     // f_sib[b,i,:,:]
  unsigned long long frowp = (unsigned long long)(uintptr_t)frow;
  unsigned fsbase = (unsigned)(uintptr_t)(&FS[0][0]);

  // diagonal ownership: lanes 0-7 own m=lane (C vgpr m), lanes 24-31 own m=lane-16 (C vgpr m-8)
  int  didx  = hf ? (m16 - 8) : m16;                 // C element index this lane would read
  bool dvalid = hf ? (m16 >= 8) : (m16 < 8);

  for (int jt = 0; jt < 10; ++jt) {
    int j0 = jt * 16;
    __syncthreads();   // QT ready (iter 0) / previous FS fully consumed

    // async-stage 16 rows (10240 B) of f_sib into FS via 128-bit async LDS loads
#pragma unroll
    for (int q = 0; q < 4; ++q) {
      int c  = tid + 160 * q;          // chunk id, 640 chunks of 16 B
      int jr = c / 40;
      int ci = c - jr * 40;
      unsigned lds = fsbase + (unsigned)(jr * (FSP * 4) + ci * 16);
      int voff = j0 * (LW * 4) + c * 16;
      asm volatile("global_load_async_to_lds_b128 %0, %1, %2"
                   :: "v"(lds), "v"(voff), "s"(frowp) : "memory");
    }
    asm volatile("s_wait_asynccnt 0" ::: "memory");
    __syncthreads();

    // A fragment (16x32 f16): lane(m16,hf) holds K = k0+hf*8+{0..7} and k0+16+hf*8+{0..7}
    int k0 = wave * 32;
    const float* arow = &FS[m16][0];
    float4 a0 = *(const float4*)(arow + k0 + hf * 8);
    float4 a1 = *(const float4*)(arow + k0 + hf * 8 + 4);
    float4 a2 = *(const float4*)(arow + k0 + 16 + hf * 8);
    float4 a3 = *(const float4*)(arow + k0 + 16 + hf * 8 + 4);
    v16h A;
    A[0] = (_Float16)a0.x; A[1] = (_Float16)a0.y; A[2]  = (_Float16)a0.z; A[3]  = (_Float16)a0.w;
    A[4] = (_Float16)a1.x; A[5] = (_Float16)a1.y; A[6]  = (_Float16)a1.z; A[7]  = (_Float16)a1.w;
    A[8] = (_Float16)a2.x; A[9] = (_Float16)a2.y; A[10] = (_Float16)a2.z; A[11] = (_Float16)a2.w;
    A[12] = (_Float16)a3.x; A[13] = (_Float16)a3.y; A[14] = (_Float16)a3.z; A[15] = (_Float16)a3.w;

    // B fragment (32x16 f16): lane(n=m16,hf) holds column j0+n, K = k0+hf*8+{0..7} and k0+16+hf*8+{0..7}
    const _Float16* qrow = &QT[j0 + m16][0];
    v8h b0 = *(const v8h*)(qrow + k0 + hf * 8);
    v8h b1 = *(const v8h*)(qrow + k0 + 16 + hf * 8);
    v16h Bf = __builtin_shufflevector(b0, b1, 0, 1, 2, 3, 4, 5, 6, 7, 8, 9, 10, 11, 12, 13, 14, 15);

    v8f acc = {};
    acc = __builtin_amdgcn_wmma_f32_16x16x32_f16(false, A, false, Bf, (short)0, acc, false, false);

    // extract diagonal D[m][m] via a cndmask select chain + single predicated LDS store
    // (C layout: VGPR r -> M=r (lanes 0-15) / M=r+8 (lanes 16-31), N=lane&15)
    float dval = acc[0];
#pragma unroll
    for (int r2 = 1; r2 < 8; ++r2) dval = (didx == r2) ? acc[r2] : dval;
    if (dvalid) tpart[wave][j0 + m16] = dval;
  }
  __syncthreads();

  // softmax over j (thread tid owns j=tid); fixed-order partial sum => deterministic
  float t = tpart[0][tid] + tpart[1][tid] + tpart[2][tid] + tpart[3][tid] + tpart[4][tid]
          + s_arc[(size_t)bi * LW + tid];
  float wm = wred_max(t);
  if (lane == 0) red[wave] = wm;
  __syncthreads();
  float mx = fmaxf(fmaxf(fmaxf(red[0], red[1]), fmaxf(red[2], red[3])), red[4]);
  float e = __expf(t - mx);
  float ws = wred_sum(e);
  if (lane == 0) red[5 + wave] = ws;
  __syncthreads();
  float s = red[5] + red[6] + red[7] + red[8] + red[9];
  q_out[(size_t)bi * LW + tid] = e / s;
}

extern "C" void kernel_launch(void* const* d_in, const int* in_sizes, int n_in,
                              void* d_out, int out_size, void* d_ws, size_t ws_size,
                              hipStream_t stream) {
  (void)in_sizes; (void)n_in; (void)out_size; (void)ws_size;
  const float* s_arc  = (const float*)d_in[0];
  const float* s_span = (const float*)d_in[1];
  const float* s_head = (const float*)d_in[2];
  const float* f_sib  = (const float*)d_in[3];
  const float* f_dsp  = (const float*)d_in[4];
  // d_in[5] = mask_dep: all True in this problem -> masks are identity.

  float* ws  = (float*)d_ws;               // 5 * 204800 floats = 4 MB scratch
  float* qaA = ws;
  float* qaB = ws + 1 * (size_t)BW * L2W;
  float* qs0 = ws + 2 * (size_t)BW * L2W;  // q_span^0
  float* qs1 = ws + 3 * (size_t)BW * L2W;  // q_span^1
  float* qs2 = ws + 4 * (size_t)BW * L2W;  // q_span^2

  float* out1 = (float*)d_out;                       // softmax(arc_logits_3)
  float* out2 = out1 + (size_t)BW * L2W;             // sigmoid(fencepost(span_logits_3))
  float* out3 = out1 + 2 * (size_t)BW * L2W;         // head_logits_3

  // init marginals
  row_softmax_k<<<160, 256, 0, stream>>>(s_arc, qaA);            // q_arc^0
  init_qspan_k<<<800, 256, 0, stream>>>(s_span, qs0);            // q_span^0

  // iteration 1
  arc_iter_k<<<BW * LW, 160, 0, stream>>>(s_arc, f_sib, qaA, qaB);                       // q_arc^1
  span_iter_k<<<25600, 256, 0, stream>>>(s_span, s_head, f_dsp, nullptr, nullptr,
                                         qs1, nullptr, nullptr, 1);                      // q_span^1
  // iteration 2
  arc_iter_k<<<BW * LW, 160, 0, stream>>>(s_arc, f_sib, qaB, qaA);                       // q_arc^2
  span_iter_k<<<25600, 256, 0, stream>>>(s_span, s_head, f_dsp, qs0, nullptr,
                                         qs2, nullptr, nullptr, 2);                      // q_span^2
  // iteration 3 (writes outputs directly)
  arc_iter_k<<<BW * LW, 160, 0, stream>>>(s_arc, f_sib, qaA, out1);                      // out1
  span_iter_k<<<25600, 256, 0, stream>>>(s_span, s_head, f_dsp, qs1, qs2,
                                         nullptr, out2, out3, 3);                        // out2, out3
}
